// RiskGNN_46411416600826
// MI455X (gfx1250) — matrix operand order repaired and checked by reading, not compiled, not run on hardware
//
#include <hip/hip_runtime.h>
#include <hip/hip_bf16.h>

typedef _Float16 half_t;
typedef __attribute__((ext_vector_type(16))) _Float16 v16h;
typedef __attribute__((ext_vector_type(8)))  _Float16 v8h;
typedef __attribute__((ext_vector_type(8)))  float    v8f;

// ---------------------------------------------------------------------------
// f32 -> f16 conversion
// ---------------------------------------------------------------------------
__global__ __launch_bounds__(256) void f32_to_f16_kernel(
    const float* __restrict__ in, half_t* __restrict__ out, int n) {
  int i = blockIdx.x * blockDim.x + threadIdx.x;
  if (i < n) out[i] = (half_t)in[i];
}

// ---------------------------------------------------------------------------
// Fused NNConv edge-message kernel.
// Per 16-edge tile (one wave):
//   h1 = relu(edge_attr @ w1 + b1)            (per-lane VALU, f32)
//   msg = (h1 (x) x_src) @ w2_flat + x_src@B2 (WMMA f16, K=1024+32, f32 acc)
//   atomic scatter-add msg into sums[dst], counts into cnt[dst]
// w2 is pre-swizzled into LDS in exact B-fragment layout (64 KB).
// ---------------------------------------------------------------------------
__global__ __launch_bounds__(256) void edge_msg_kernel(
    const float* __restrict__ edge_attr,
    const int* __restrict__ srcI, const int* __restrict__ dstI,
    const half_t* __restrict__ xh,
    const float* __restrict__ w1, const float* __restrict__ b1,
    const float* __restrict__ w2, const float* __restrict__ b2,
    float* __restrict__ sums, float* __restrict__ cnt, int E)
{
  __shared__ __attribute__((aligned(32))) half_t w2frag[32 * 1024];  // 64 KB

  // Preload + swizzle w2 into B-fragment layout:
  // frag element j of lane n, k-block k, n-tile t  <-  w2[k*1024 + i*32 + o]
  //   i = j + 16*(n>=16), o = (n&15) + 16*t
  for (int f = threadIdx.x; f < 32 * 1024; f += blockDim.x) {
    int k = f >> 10, rem = f & 1023;
    int t = rem >> 9, n = (rem >> 4) & 31, j = rem & 15;
    int i = j + ((n >> 4) << 4);
    int o = (n & 15) + (t << 4);
    w2frag[f] = (half_t)w2[k * 1024 + i * 32 + o];
  }
  __syncthreads();

  int lane  = threadIdx.x & 31;
  int hi    = lane >> 4;     // 0: lanes 0-15, 1: lanes 16-31
  int laneM = lane & 15;

  // Bias fragments (extra K-block with A = x_frag unscaled): B[i,o] = b2[i*32+o]
  v16h bb0, bb1;
  #pragma unroll
  for (int j = 0; j < 16; ++j) {
    int i = j + hi * 16;
    bb0[j] = (half_t)b2[i * 32 + laneM];
    bb1[j] = (half_t)b2[i * 32 + laneM + 16];
  }

  int waveId   = (blockIdx.x * blockDim.x + threadIdx.x) >> 5;
  int numWaves = (gridDim.x * blockDim.x) >> 5;
  int numTiles = (E + 15) >> 4;

  for (int tile = waveId; tile < numTiles; tile += numWaves) {
    int  e     = tile * 16 + laneM;  // lanes L and L+16 handle the same edge
    bool valid = e < E;
    int  s = 0, d = 0;
    float ea[16];
    #pragma unroll
    for (int c = 0; c < 16; ++c) ea[c] = 0.0f;
    if (valid) {
      s = srcI[e];
      d = dstI[e];
      #pragma unroll
      for (int c = 0; c < 16; ++c) ea[c] = edge_attr[e * 16 + c];
    }

    // Edge MLP layer 1: h1[k] = relu(ea . w1[:,k] + b1[k])  (w1 loads are
    // wave-uniform -> scalar loads)
    float h1[32];
    #pragma unroll
    for (int k = 0; k < 32; ++k) {
      float acc = b1[k];
      #pragma unroll
      for (int c = 0; c < 16; ++c) acc += ea[c] * w1[c * 32 + k];
      h1[k] = fmaxf(acc, 0.0f);
    }

    // x_src fragment as WMMA A operand, contiguous k order (same as B):
    //   lane<16: i in {0..15}; lane>=16: i in {16..31}
    const half_t* xr = xh + s * 32 + hi * 16;
    v8h lo = *(const v8h*)(xr);
    v8h hv = *(const v8h*)(xr + 8);
    v16h xf;
    #pragma unroll
    for (int j = 0; j < 8; ++j) { xf[j] = lo[j]; xf[j + 8] = hv[j]; }

    v8f c0 = {}, c1 = {};
    #pragma unroll 4
    for (int k = 0; k < 32; ++k) {
      half_t hk = (half_t)h1[k];
      v16h a;
      #pragma unroll
      for (int j = 0; j < 16; ++j) a[j] = xf[j] * hk;   // packed f16 scale
      v16h bf0 = *(const v16h*)&w2frag[k * 1024 + lane * 16];
      v16h bf1 = *(const v16h*)&w2frag[k * 1024 + 512 + lane * 16];
      c0 = __builtin_amdgcn_wmma_f32_16x16x32_f16(false, a, false, bf0,
                                                  (short)0, c0, false, false);
      c1 = __builtin_amdgcn_wmma_f32_16x16x32_f16(false, a, false, bf1,
                                                  (short)0, c1, false, false);
    }
    // bias block: A = x_frag, B = b2 fragments
    c0 = __builtin_amdgcn_wmma_f32_16x16x32_f16(false, xf, false, bb0,
                                                (short)0, c0, false, false);
    c1 = __builtin_amdgcn_wmma_f32_16x16x32_f16(false, xf, false, bb1,
                                                (short)0, c1, false, false);

    if (valid && hi == 0) atomicAdd(&cnt[d], 1.0f);

    // C/D layout: lane n holds col o=(n&15); VGPR r holds row M = r + 8*hi
    #pragma unroll
    for (int r = 0; r < 8; ++r) {
      int M  = r + hi * 8;
      int dM = __shfl(d, M, 32);          // lane M holds edge M's dst
      bool vM = (tile * 16 + M) < E;
      if (vM) {
        atomicAdd(&sums[dM * 32 + laneM],      c0[r]);
        atomicAdd(&sums[dM * 32 + laneM + 16], c1[r]);
      }
    }
  }
}

// ---------------------------------------------------------------------------
// Node update: h = relu(sums/max(cnt,1) + x@root + bias); write f32 + f16 copy
// ---------------------------------------------------------------------------
__global__ __launch_bounds__(256) void node_update_kernel(
    const float* __restrict__ sums, const float* __restrict__ cnt,
    const float* __restrict__ xin, const float* __restrict__ root,
    const float* __restrict__ bias, float* __restrict__ hout,
    half_t* __restrict__ houth, int N)
{
  int t = blockIdx.x * blockDim.x + threadIdx.x;
  if (t >= N * 32) return;
  int n = t >> 5, o = t & 31;
  float c   = fmaxf(cnt[n], 1.0f);
  float acc = sums[t] / c + bias[o];
  #pragma unroll
  for (int i = 0; i < 32; ++i) acc += xin[(n << 5) + i] * root[(i << 5) + o];
  float v = fmaxf(acc, 0.0f);
  hout[t]  = v;
  houth[t] = (half_t)v;
}

// ---------------------------------------------------------------------------
// Action head: one wave per action. inp = [att_e(32), def_e(32), ns] (65)
// hidden 128 = 4 per lane; wave-reduce the final dot.
// ---------------------------------------------------------------------------
__global__ __launch_bounds__(256) void action_head_kernel(
    const int* __restrict__ table, const float* __restrict__ emb,
    const float* __restrict__ skipA, const float* __restrict__ skipD,
    const float* __restrict__ w1, const float* __restrict__ b1,
    const float* __restrict__ w2, const float* __restrict__ b2,
    float* __restrict__ out, int A)
{
  int gt   = blockIdx.x * blockDim.x + threadIdx.x;
  int a    = gt >> 5;
  int lane = threadIdx.x & 31;
  if (a >= A) return;  // wave-uniform

  int   att = table[a * 3 + 0];
  int   def = table[a * 3 + 1];
  float nsf = (float)table[a * 3 + 2];
  bool  skip = (att == -1);
  float inA = skip ? skipA[lane] : emb[att * 32 + lane];
  float inD = skip ? skipD[lane] : emb[def * 32 + lane];

  float hs0 = b1[lane], hs1 = b1[lane + 32], hs2 = b1[lane + 64], hs3 = b1[lane + 96];
  for (int c = 0; c < 32; ++c) {
    float av = __shfl(inA, c, 32);
    float dv = __shfl(inD, c, 32);
    const float* wr  = w1 + c * 128 + lane;
    const float* wr2 = w1 + (c + 32) * 128 + lane;
    hs0 += av * wr[0]  + dv * wr2[0];
    hs1 += av * wr[32] + dv * wr2[32];
    hs2 += av * wr[64] + dv * wr2[64];
    hs3 += av * wr[96] + dv * wr2[96];
  }
  const float* wn = w1 + 64 * 128 + lane;
  hs0 += nsf * wn[0]; hs1 += nsf * wn[32]; hs2 += nsf * wn[64]; hs3 += nsf * wn[96];

  float acc = fmaxf(hs0, 0.f) * w2[lane]      + fmaxf(hs1, 0.f) * w2[lane + 32]
            + fmaxf(hs2, 0.f) * w2[lane + 64] + fmaxf(hs3, 0.f) * w2[lane + 96];
  #pragma unroll
  for (int off = 16; off; off >>= 1) acc += __shfl_down(acc, off, 32);
  if (lane == 0) out[a] = acc + b2[0];
}

// ---------------------------------------------------------------------------
extern "C" void kernel_launch(void* const* d_in, const int* in_sizes, int n_in,
                              void* d_out, int out_size, void* d_ws, size_t ws_size,
                              hipStream_t stream)
{
  const float* x         = (const float*)d_in[0];
  const int*   eidx      = (const int*)d_in[1];
  const float* edge_attr = (const float*)d_in[2];
  const int*   table     = (const int*)d_in[3];
  const float* en1_w1 = (const float*)d_in[4];
  const float* en1_b1 = (const float*)d_in[5];
  const float* en1_w2 = (const float*)d_in[6];
  const float* en1_b2 = (const float*)d_in[7];
  const float* root1  = (const float*)d_in[8];
  const float* bias1  = (const float*)d_in[9];
  const float* en2_w1 = (const float*)d_in[10];
  const float* en2_b1 = (const float*)d_in[11];
  const float* en2_w2 = (const float*)d_in[12];
  const float* en2_b2 = (const float*)d_in[13];
  const float* root2  = (const float*)d_in[14];
  const float* bias2  = (const float*)d_in[15];
  const float* skipA  = (const float*)d_in[16];
  const float* skipD  = (const float*)d_in[17];
  const float* mlp_w1 = (const float*)d_in[18];
  const float* mlp_b1 = (const float*)d_in[19];
  const float* mlp_w2 = (const float*)d_in[20];
  const float* mlp_b2 = (const float*)d_in[21];
  float* out = (float*)d_out;

  int N = in_sizes[0] / 32;
  int E = in_sizes[1] / 2;
  int A = in_sizes[3] / 3;
  const int* srcI = eidx;
  const int* dstI = eidx + E;

  size_t off = 0;
  auto take = [&](size_t bytes) {
    char* p = (char*)d_ws + off;
    off += (bytes + 255) & ~(size_t)255;
    return (void*)p;
  };
  half_t* xh   = (half_t*)take((size_t)N * 32 * sizeof(half_t));
  float*  sums = (float*) take((size_t)N * 32 * sizeof(float));
  float*  cnt  = (float*) take((size_t)N * sizeof(float));
  float*  h1n  = (float*) take((size_t)N * 32 * sizeof(float));
  half_t* h1h  = (half_t*)take((size_t)N * 32 * sizeof(half_t));
  float*  h2n  = (float*) take((size_t)N * 32 * sizeof(float));
  (void)ws_size; (void)n_in; (void)out_size;

  int nv = N * 32;
  f32_to_f16_kernel<<<(nv + 255) / 256, 256, 0, stream>>>(x, xh, nv);

  // Layer 1
  hipMemsetAsync(sums, 0, (size_t)nv * 4, stream);
  hipMemsetAsync(cnt,  0, (size_t)N * 4, stream);
  edge_msg_kernel<<<256, 256, 0, stream>>>(edge_attr, srcI, dstI, xh,
                                           en1_w1, en1_b1, en1_w2, en1_b2,
                                           sums, cnt, E);
  node_update_kernel<<<(nv + 255) / 256, 256, 0, stream>>>(sums, cnt, x, root1,
                                                           bias1, h1n, h1h, N);
  // Layer 2
  hipMemsetAsync(sums, 0, (size_t)nv * 4, stream);
  hipMemsetAsync(cnt,  0, (size_t)N * 4, stream);
  edge_msg_kernel<<<256, 256, 0, stream>>>(edge_attr, srcI, dstI, h1h,
                                           en2_w1, en2_b1, en2_w2, en2_b2,
                                           sums, cnt, E);
  node_update_kernel<<<(nv + 255) / 256, 256, 0, stream>>>(sums, cnt, h1n, root2,
                                                           bias2, h2n, xh /*scratch f16*/, N);
  // Action head
  action_head_kernel<<<((size_t)A * 32 + 255) / 256, 256, 0, stream>>>(
      table, h2n, skipA, skipD, mlp_w1, mlp_b1, mlp_w2, mlp_b2, out, A);
}
